// WindowAttentionLayer_19653770346959
// MI455X (gfx1250) — compile-verified, matrix operand order checked by reading, not run on hardware
//
#include <hip/hip_runtime.h>
#include <hip/hip_bf16.h>

typedef _Float16 f16;
typedef __attribute__((ext_vector_type(16))) _Float16 f16x16;
typedef __attribute__((ext_vector_type(8)))  float    f32x8;

union Frag { f16x16 v; uint4 q[2]; };

// Load a 16x32 f16 fragment in the CDNA5 WMMA A/B register layout from a
// row-major [R][stride] array (rows = M for A, rows = N for B since our B
// operands are all stored as [N][K]).
// Lane L in [0,16): row = row0+L,   holds k0+0..7 (v0..3) and k0+16..23 (v4..7)
// Lane L in [16,32): row = row0+L-16, holds k0+8..15 and k0+24..31
__device__ __forceinline__ f16x16 ldfrag(const f16* base, int row0, int stride, int k0) {
  int lane = threadIdx.x & 31;
  const f16* p = base + (row0 + (lane & 15)) * stride + k0 + ((lane >> 4) << 3);
  Frag f;
  f.q[0] = *(const uint4*)p;         // k .. k+7
  f.q[1] = *(const uint4*)(p + 16);  // k+16 .. k+23
  return f.v;
}

__device__ __forceinline__ f32x8 wmma16(f16x16 a, f16x16 b, f32x8 c) {
  return __builtin_amdgcn_wmma_f32_16x16x32_f16(false, a, false, b, (short)0, c, false, false);
}

// ---- problem constants ----
#define BT    128
#define HW    32
#define DIM   512
#define WSZ   8
#define SSZ   4
#define NHD   8
#define NTOK  64        // tokens per window
// LDS strides (elements), padded to rotate banks
#define LXS   520       // X / O  f16 [64][520]
#define LSS   68        // S      f32 [64][68]
#define LVS   72        // VT     f16 [8][64][72] ; P f16 [64][72]
#define OSS   516       // out staging f32 [64][516]
// LDS byte offsets
#define OFF_X   0
#define OFF_Q   66560
#define OFF_K   133120
#define OFF_VT  199680
#define OFF_S   273408
#define OFF_P   290816
#define SMEM_BYTES 300032
#define OFF_OUTS OFF_Q   // reuses dead Q+K (needs 132096 <= 133120)

__global__ void __launch_bounds__(256)
cvt_f16_kernel(const float* __restrict__ src, f16* __restrict__ dst, int n) {
  int i = (blockIdx.x * 256 + threadIdx.x) * 4;
  if (i < n) {
    float4 f = *(const float4*)(src + i);
    f16 o[4] = { (f16)f.x, (f16)f.y, (f16)f.z, (f16)f.w };
    *(uint2*)(dst + i) = *(const uint2*)o;
  }
}

__global__ void __launch_bounds__(256)
swin_window_attn_kernel(const float* __restrict__ x,
                        const f16*   __restrict__ qkvw,   // [1536][512] f16
                        const float* __restrict__ qkvb,   // [1536]
                        const f16*   __restrict__ projw,  // [512][512] f16
                        const float* __restrict__ projb,  // [512]
                        const float* __restrict__ rpb,    // [225][8]
                        float* __restrict__ out) {
  extern __shared__ char smem[];
  f16*   X    = (f16*)(smem + OFF_X);    // input window, later O
  f16*   Q    = (f16*)(smem + OFF_Q);
  f16*   K    = (f16*)(smem + OFF_K);
  f16*   VT   = (f16*)(smem + OFF_VT);   // [head][dim][token]
  float* S    = (float*)(smem + OFF_S);
  f16*   P    = (f16*)(smem + OFF_P);
  float* OUTS = (float*)(smem + OFF_OUTS);

  const int tid  = threadIdx.x;
  const int lane = tid & 31;
  const int wv   = tid >> 5;
  const int widx = blockIdx.x;
  const int bt = widx >> 4;
  const int wh = (widx >> 2) & 3;
  const int ww = widx & 3;
  const int laneR = lane & 15;
  const int laneG = (lane >> 4) << 3;

  // ---- 1) gather cyclically-shifted window -> X f16 ----
  for (int t8 = 0; t8 < 8; ++t8) {
    int m = wv * 8 + t8;
    int r = m >> 3, c = m & 7;
    int gr = (wh * 8 + r + SSZ) & 31;
    int gc = (ww * 8 + c + SSZ) & 31;
    const float* src = x + (((size_t)bt * HW + gr) * HW + gc) * DIM + lane * 16;
    f16* dst = X + m * LXS + lane * 16;
#pragma unroll
    for (int q4 = 0; q4 < 4; ++q4) {
      float4 f = *(const float4*)(src + q4 * 4);
      f16 o[4] = { (f16)f.x, (f16)f.y, (f16)f.z, (f16)f.w };
      *(uint2*)(dst + q4 * 4) = *(const uint2*)o;
    }
  }
  __syncthreads();

  // ---- 2) QKV GEMM: [64,512] x [512,1536]^T, 4x96 tiles, 8 waves * 6 passes ----
  for (int pass = 0; pass < 6; ++pass) {
    int gid = wv + 8 * pass;          // 0..47
    int m0 = (gid & 1) * 32;
    int n0 = (gid >> 1) * 64;
    f32x8 acc[2][4] = {};
    for (int ks = 0; ks < 16; ++ks) {
      int k0 = ks * 32;
      f16x16 a0 = ldfrag(X, m0,      LXS, k0);
      f16x16 a1 = ldfrag(X, m0 + 16, LXS, k0);
#pragma unroll
      for (int j = 0; j < 4; ++j) {
        f16x16 b = ldfrag(qkvw, n0 + j * 16, DIM, k0);
        acc[0][j] = wmma16(a0, b, acc[0][j]);
        acc[1][j] = wmma16(a1, b, acc[1][j]);
      }
    }
#pragma unroll
    for (int i = 0; i < 2; ++i) {
      int mb = m0 + i * 16 + laneG;
#pragma unroll
      for (int j = 0; j < 4; ++j) {
        int n = n0 + j * 16 + laneR;
        float bia = qkvb[n];
        if (n0 < 512) {                       // Q, pre-scaled by hd^-0.5
#pragma unroll
          for (int e = 0; e < 8; ++e)
            Q[(mb + e) * LXS + n] = (f16)((acc[i][j][e] + bia) * 0.125f);
        } else if (n0 < 1024) {               // K
          int d = n - 512;
#pragma unroll
          for (int e = 0; e < 8; ++e)
            K[(mb + e) * LXS + d] = (f16)(acc[i][j][e] + bia);
        } else {                              // V, stored transposed [dim][token]
          int d = n - 1024, hh = d >> 6, dd = d & 63;
          f16 o[8];
#pragma unroll
          for (int e = 0; e < 8; ++e) o[e] = (f16)(acc[i][j][e] + bia);
          *(uint4*)(VT + (hh * 64 + dd) * LVS + mb) = *(const uint4*)o;
        }
      }
    }
  }
  __syncthreads();

  // ---- 3) attention per head ----
  for (int h = 0; h < NHD; ++h) {
    // S = Q Kt : 16 tiles, 2 per wave
#pragma unroll
    for (int tt = 0; tt < 2; ++tt) {
      int tile = wv * 2 + tt;
      int mi = tile >> 2, ni = tile & 3;
      f32x8 acc = {};
#pragma unroll
      for (int ks = 0; ks < 2; ++ks) {
        int k0 = h * 64 + ks * 32;
        f16x16 a = ldfrag(Q, mi * 16, LXS, k0);
        f16x16 b = ldfrag(K, ni * 16, LXS, k0);
        acc = wmma16(a, b, acc);
      }
      int n = ni * 16 + laneR;
      int mb = mi * 16 + laneG;
#pragma unroll
      for (int e = 0; e < 8; ++e) S[(mb + e) * LSS + n] = acc[e];
    }
    __syncthreads();

    // softmax over rows + rel-pos bias + shift mask (4 threads per row)
    {
      int row = tid >> 2, part = tid & 3;
      int ri = row >> 3, ci = row & 7;
      int cr = ((wh == 3) ? (ri < 4 ? 1 : 2) * 3 : 0) + ((ww == 3) ? (ci < 4 ? 1 : 2) : 0);
      float vals[16];
      float mx = -1e30f;
#pragma unroll
      for (int e = 0; e < 16; ++e) {
        int col = part * 16 + e;
        int rj = col >> 3, cj = col & 7;
        int cc = ((wh == 3) ? (rj < 4 ? 1 : 2) * 3 : 0) + ((ww == 3) ? (cj < 4 ? 1 : 2) : 0);
        float bia = rpb[((ri - rj + 7) * 15 + (ci - cj + 7)) * NHD + h];
        float v = S[row * LSS + col] + bia + ((cr != cc) ? -100.0f : 0.0f);
        vals[e] = v;
        mx = fmaxf(mx, v);
      }
      mx = fmaxf(mx, __shfl_xor(mx, 1, 32));
      mx = fmaxf(mx, __shfl_xor(mx, 2, 32));
      float sum = 0.0f;
#pragma unroll
      for (int e = 0; e < 16; ++e) { vals[e] = __expf(vals[e] - mx); sum += vals[e]; }
      sum += __shfl_xor(sum, 1, 32);
      sum += __shfl_xor(sum, 2, 32);
      float inv = 1.0f / sum;
      f16 o[16];
#pragma unroll
      for (int e = 0; e < 16; ++e) o[e] = (f16)(vals[e] * inv);
      *(uint4*)(P + row * LVS + part * 16)     = ((const uint4*)o)[0];
      *(uint4*)(P + row * LVS + part * 16 + 8) = ((const uint4*)o)[1];
    }
    __syncthreads();

    // O = P V : B operand is VT[h] ([dim][token] == V^T), same fragment loader
#pragma unroll
    for (int tt = 0; tt < 2; ++tt) {
      int tile = wv * 2 + tt;
      int mi = tile >> 2, ni = tile & 3;
      f32x8 acc = {};
#pragma unroll
      for (int ks = 0; ks < 2; ++ks) {
        f16x16 a = ldfrag(P, mi * 16, LVS, ks * 32);
        f16x16 b = ldfrag(VT + h * 64 * LVS, ni * 16, LVS, ks * 32);
        acc = wmma16(a, b, acc);
      }
      int n = h * 64 + ni * 16 + laneR;
      int mb = mi * 16 + laneG;
#pragma unroll
      for (int e = 0; e < 8; ++e) X[(mb + e) * LXS + n] = (f16)acc[e];  // X reused as O
    }
    __syncthreads();
  }

  // ---- 4) proj GEMM: [64,512] x [512,512]^T, 4x32 tiles, 8 waves * 2 passes ----
  for (int pass = 0; pass < 2; ++pass) {
    int gid = wv + 8 * pass;          // 0..15
    int m0 = (gid & 1) * 32;
    int n0 = (gid >> 1) * 64;
    f32x8 acc[2][4] = {};
    for (int ks = 0; ks < 16; ++ks) {
      int k0 = ks * 32;
      f16x16 a0 = ldfrag(X, m0,      LXS, k0);
      f16x16 a1 = ldfrag(X, m0 + 16, LXS, k0);
#pragma unroll
      for (int j = 0; j < 4; ++j) {
        f16x16 b = ldfrag(projw, n0 + j * 16, DIM, k0);
        acc[0][j] = wmma16(a0, b, acc[0][j]);
        acc[1][j] = wmma16(a1, b, acc[1][j]);
      }
    }
#pragma unroll
    for (int i = 0; i < 2; ++i) {
      int mb = m0 + i * 16 + laneG;
#pragma unroll
      for (int j = 0; j < 4; ++j) {
        int n = n0 + j * 16 + laneR;
        float bia = projb[n];
#pragma unroll
        for (int e = 0; e < 8; ++e)
          OUTS[(mb + e) * OSS + n] = acc[i][j][e] + bia;
      }
    }
  }
  __syncthreads();

  // ---- 5) coalesced scatter with reverse cyclic shift ----
  for (int t8 = 0; t8 < 8; ++t8) {
    int m = wv * 8 + t8;
    int r = m >> 3, c = m & 7;
    int gr = (wh * 8 + r + SSZ) & 31;
    int gc = (ww * 8 + c + SSZ) & 31;
    float* dst = out + (((size_t)bt * HW + gr) * HW + gc) * DIM + lane * 16;
    const float* src = OUTS + m * OSS + lane * 16;
#pragma unroll
    for (int q4 = 0; q4 < 4; ++q4)
      *(float4*)(dst + q4 * 4) = *(const float4*)(src + q4 * 4);
  }
}

extern "C" void kernel_launch(void* const* d_in, const int* in_sizes, int n_in,
                              void* d_out, int out_size, void* d_ws, size_t ws_size,
                              hipStream_t stream) {
  const float* x      = (const float*)d_in[0];
  const float* qkvw_f = (const float*)d_in[1];
  const float* qkvb   = (const float*)d_in[2];
  const float* projw_f= (const float*)d_in[3];
  const float* projb  = (const float*)d_in[4];
  const float* rpb    = (const float*)d_in[5];
  float* out = (float*)d_out;

  const int QKVW_N = 3 * DIM * DIM;   // 786432
  const int PROJW_N = DIM * DIM;      // 262144
  f16* qkvw_h  = (f16*)d_ws;
  f16* projw_h = (f16*)((char*)d_ws + (size_t)QKVW_N * sizeof(f16));

  cvt_f16_kernel<<<(QKVW_N / 4 + 255) / 256, 256, 0, stream>>>(qkvw_f, qkvw_h, QKVW_N);
  cvt_f16_kernel<<<(PROJW_N / 4 + 255) / 256, 256, 0, stream>>>(projw_f, projw_h, PROJW_N);

  (void)hipFuncSetAttribute((const void*)swin_window_attn_kernel,
                            hipFuncAttributeMaxDynamicSharedMemorySize, SMEM_BYTES);
  swin_window_attn_kernel<<<2048, 256, SMEM_BYTES, stream>>>(
      x, qkvw_h, qkvb, projw_h, projb, rpb, out);
}
